// MemorizingTransformer_66108136620775
// MI455X (gfx1250) — matrix operand order repaired and checked by reading, not compiled, not run on hardware
//
#include <hip/hip_runtime.h>
#include <hip/hip_bf16.h>
#include <math.h>

typedef __attribute__((ext_vector_type(16))) _Float16 v16h;
typedef __attribute__((ext_vector_type(8)))  _Float16 v8h;
typedef __attribute__((ext_vector_type(8)))  float    v8f;

// ---------------------------------------------------------------------------
// WMMA helpers (gfx1250, wave32).  D = A(16x32 f16) * B(32x16 f16) + C(16x16 f32)
// ---------------------------------------------------------------------------
__device__ __forceinline__ v8f wmma16(v16h a, v16h b, v8f c) {
  return __builtin_amdgcn_wmma_f32_16x16x32_f16(
      /*neg_a=*/false, a, /*neg_b=*/false, b,
      /*c_mod=*/(short)0, c, /*reuse_a=*/false, /*reuse_b=*/false);
}

// A-fragment (16x32, f16) from fp32 row-major A[lda], rows row0.., K-slice k0..k0+31.
// Lane L holds row (L&15); elems 0..7 -> K = k0 + (L<16?0:8) + i ; elems 8..15 -> +16.
__device__ __forceinline__ v16h afrag_f32(const float* A, int lda, int row0, int k0) {
  int lane = threadIdx.x & 31;
  const float* r = A + (size_t)(row0 + (lane & 15)) * lda + k0 + ((lane & 16) ? 8 : 0);
  v16h a;
#pragma unroll
  for (int i = 0; i < 8; ++i) {
    a[i]     = (_Float16)r[i];
    a[i + 8] = (_Float16)r[16 + i];
  }
  return a;
}

// Same A-fragment but from f16 row-major storage (two aligned 16B loads).
__device__ __forceinline__ v16h afrag_f16(const _Float16* A, int lda, int row0, int k0) {
  int lane = threadIdx.x & 31;
  const _Float16* r = A + (size_t)(row0 + (lane & 15)) * lda + k0 + ((lane & 16) ? 8 : 0);
  v8h lo = *(const v8h*)r;
  v8h hi = *(const v8h*)(r + 16);
  v16h a;
#pragma unroll
  for (int i = 0; i < 8; ++i) { a[i] = lo[i]; a[i + 8] = hi[i]; }
  return a;
}

// B-fragment (32x16, f16) from fp32 row-major B[K][N] (weights).
// Lane L holds col (L&15); elems 0..7 -> K = k0 + (L<16?0:8) + i ; elems 8..15 -> +16.
__device__ __forceinline__ v16h bfrag_f32(const float* B, int ldb, int k0, int col0) {
  int lane = threadIdx.x & 31;
  const float* p = B + (size_t)(k0 + ((lane & 16) ? 8 : 0)) * ldb + col0 + (lane & 15);
  v16h b;
#pragma unroll
  for (int i = 0; i < 16; ++i) b[i] = (_Float16)p[(size_t)(i < 8 ? i : i + 8) * ldb];
  return b;
}

// B-fragment (32x16, f16) from *transposed* f16 storage T[N][K] (row per B-column).
// Two aligned 16B loads per lane.
__device__ __forceinline__ v16h bfrag_T16(const _Float16* T, int ldt, int col0, int k0) {
  int lane = threadIdx.x & 31;
  const _Float16* p = T + (size_t)(col0 + (lane & 15)) * ldt + k0 + ((lane & 16) ? 8 : 0);
  v8h lo = *(const v8h*)p;
  v8h hi = *(const v8h*)(p + 16);
  v16h b;
#pragma unroll
  for (int i = 0; i < 8; ++i) { b[i] = lo[i]; b[i + 8] = hi[i]; }
  return b;
}

// ---------------------------------------------------------------------------
// Kernel 1: C(f32) = A(f32) * B(f32) via f16 WMMA. One wave = one 16x16 tile.
// grid.x = M/16, grid.y covers N in 4-wave strips (block=128).
// ---------------------------------------------------------------------------
__global__ __launch_bounds__(128)
void gemm_f32_wmma(const float* __restrict__ A, const float* __restrict__ B,
                   float* __restrict__ C, int K, int N) {
  int w = threadIdx.x >> 5;
  int lane = threadIdx.x & 31;
  int mt = blockIdx.x * 16;
  int nt = (blockIdx.y * 4 + w) * 16;
  v8f c = {};
  for (int k0 = 0; k0 < K; k0 += 32)
    c = wmma16(afrag_f32(A, K, mt, k0), bfrag_f32(B, N, k0, nt), c);
  int col = nt + (lane & 15);
  int rbase = (lane & 16) ? 8 : 0;
#pragma unroll
  for (int r = 0; r < 8; ++r)
    C[(size_t)(mt + rbase + r) * N + col] = c[r];
}

// ---------------------------------------------------------------------------
// Kernel 2a: l2-normalize q per (b,n,h) over dh=64, scale by exp(scale_param[h]),
// write fp32 (for mem-dots) and f16 (for WMMA) in (b,h,n,dh) layout.
// One wave per (b,n,h) row-head; each lane owns 2 elements.
// ---------------------------------------------------------------------------
__global__ __launch_bounds__(256)
void norm_q_kernel(const float* __restrict__ qraw, const float* __restrict__ scale_param,
                   float* __restrict__ qn, _Float16* __restrict__ qh) {
  const int n = 2048, H = 8;
  int gw = blockIdx.x * 8 + (threadIdx.x >> 5);
  int lane = threadIdx.x & 31;
  int bIdx = gw / (n * H);
  int rem  = gw % (n * H);
  int nIdx = rem / H;
  int hIdx = rem % H;
  const float* src = qraw + ((size_t)(bIdx * n + nIdx)) * 512 + hIdx * 64;
  float e0 = src[lane * 2], e1 = src[lane * 2 + 1];
  float ss = e0 * e0 + e1 * e1;
#pragma unroll
  for (int off = 16; off >= 1; off >>= 1) ss += __shfl_xor(ss, off, 32);
  float inv = expf(scale_param[hIdx]) / fmaxf(sqrtf(ss), 1e-12f);
  size_t o = (((size_t)(bIdx * H + hIdx) * n) + nIdx) * 64 + lane * 2;
  float v0 = e0 * inv, v1 = e1 * inv;
  qn[o] = v0; qn[o + 1] = v1;
  qh[o] = (_Float16)v0; qh[o + 1] = (_Float16)v1;
}

// ---------------------------------------------------------------------------
// Kernel 2b: l2-normalize k (no scale) -> kf16 (b,n,dh); transpose v -> vT (b,dh,n) f16.
// One wave per (b,n) row.
// ---------------------------------------------------------------------------
__global__ __launch_bounds__(256)
void norm_kv_kernel(const float* __restrict__ kvraw, _Float16* __restrict__ kf16,
                    _Float16* __restrict__ vT) {
  const int n = 2048;
  int gw = blockIdx.x * 8 + (threadIdx.x >> 5);
  int lane = threadIdx.x & 31;
  int bIdx = gw / n, nIdx = gw % n;
  const float* src = kvraw + (size_t)gw * 128;
  float e0 = src[lane * 2], e1 = src[lane * 2 + 1];
  float ss = e0 * e0 + e1 * e1;
#pragma unroll
  for (int off = 16; off >= 1; off >>= 1) ss += __shfl_xor(ss, off, 32);
  float inv = 1.0f / fmaxf(sqrtf(ss), 1e-12f);
  size_t ko = (size_t)gw * 64 + lane * 2;
  kf16[ko]     = (_Float16)(e0 * inv);
  kf16[ko + 1] = (_Float16)(e1 * inv);
  float v0 = src[64 + lane * 2], v1 = src[64 + lane * 2 + 1];
  vT[((size_t)bIdx * 64 + lane * 2) * n + nIdx]     = (_Float16)v0;
  vT[((size_t)bIdx * 64 + lane * 2 + 1) * n + nIdx] = (_Float16)v1;
}

// ---------------------------------------------------------------------------
// Kernel 3: flash attention with per-query memory keys.
// One wave = one 16-query tile of one (b,h). block = 4 waves, grid = (n/64, b*h).
// ---------------------------------------------------------------------------
__global__ __launch_bounds__(128)
void attn_kernel(const _Float16* __restrict__ qh, const float* __restrict__ qn,
                 const _Float16* __restrict__ kf16, const _Float16* __restrict__ vT,
                 const float* __restrict__ mem_kv, const unsigned char* __restrict__ mem_mask,
                 const float* __restrict__ bias, _Float16* __restrict__ attn_out) {
  const int n = 2048, dh = 64, KRET = 32;
  __shared__ __align__(32) _Float16 pbuf[4][16][32];   // P tile staging (C->A layout bounce)
  __shared__ __align__(32) float    pmbuf[4][16][32];  // mem-attn probabilities

  int w = threadIdx.x >> 5, lane = threadIdx.x & 31;
  int half = lane >> 4;                 // 0: rows 0..7, 1: rows 8..15 (C layout)
  int bh = blockIdx.y;                  // b*8 + h
  int bIdx = bh >> 3, hIdx = bh & 7;
  int qt = blockIdx.x * 4 + w;
  int qRow0 = qt * 16;

  const _Float16* qb   = qh   + (size_t)bh * n * dh;
  const float*    qnb  = qn   + (size_t)bh * n * dh;
  const _Float16* kb   = kf16 + (size_t)bIdx * n * dh;
  const _Float16* vb   = vT   + (size_t)bIdx * dh * n;
  const float*    bb   = bias + (size_t)hIdx * n * n;

  // q A-fragments (dh = 64 -> two K-slices of 32), loaded once.
  v16h a0 = afrag_f16(qb, dh, qRow0, 0);
  v16h a1 = afrag_f16(qb, dh, qRow0, 32);

  // ---- memory-key scores: lane j owns mem slot j for every query row m ----
  float smem_s[16];
  float rowM[8], rowL[8];
#pragma unroll
  for (int m = 0; m < 16; ++m) {
    int i = qRow0 + m;
    const float* mk = mem_kv + ((((size_t)bh * n + i) * KRET + lane) * 2 + 0) * dh;
    const float* qv = qnb + (size_t)i * dh;
    float s = 0.f;
#pragma unroll 8
    for (int d = 0; d < 64; ++d) s = fmaf(qv[d], mk[d], s);
    if (!mem_mask[((size_t)bh * n + i) * KRET + lane]) s = -1e30f;
    smem_s[m] = s;
    float mx = s;
#pragma unroll
    for (int off = 16; off >= 1; off >>= 1) mx = fmaxf(mx, __shfl_xor(mx, off, 32));
    float p = (s > -1e29f) ? expf(s - mx) : 0.f;
    float sm = p;
#pragma unroll
    for (int off = 16; off >= 1; off >>= 1) sm += __shfl_xor(sm, off, 32);
    if ((m >> 3) == half) { rowM[m & 7] = mx; rowL[m & 7] = sm; }
  }

  v8f o[4];
#pragma unroll
  for (int t = 0; t < 4; ++t) o[t] = v8f{};
  v8f z = {};

  // ---- causal flash loop over 32 keys per step ----
  int kEnd = qRow0 + 16;
  for (int k0 = 0; k0 < kEnd; k0 += 32) {
    if (k0 + 32 < kEnd) {
      __builtin_prefetch(kb + (size_t)(k0 + 32) * dh, 0, 0);     // global_prefetch_b8
      __builtin_prefetch(vb + (size_t)(k0 + 32), 0, 0);
    }
    // S tiles: keys [k0,k0+16) and [k0+16,k0+32); chain WMMAs across dh slices.
    v8f s_lo = wmma16(a1, bfrag_T16(kb, dh, k0,      32), wmma16(a0, bfrag_T16(kb, dh, k0,      0), z));
    v8f s_hi = wmma16(a1, bfrag_T16(kb, dh, k0 + 16, 32), wmma16(a0, bfrag_T16(kb, dh, k0 + 16, 0), z));

    int colL = k0 + (lane & 15);
    int colH = colL + 16;
    v8f tmax;
#pragma unroll
    for (int r = 0; r < 8; ++r) {
      int i = qRow0 + half * 8 + r;
      float sl = s_lo[r] + bb[(size_t)i * n + colL];
      float sh = s_hi[r] + bb[(size_t)i * n + colH];
      if (colL > i) sl = -1e30f;
      if (colH > i) sh = -1e30f;
      s_lo[r] = sl; s_hi[r] = sh;
      tmax[r] = fmaxf(sl, sh);
    }
#pragma unroll
    for (int off = 8; off >= 1; off >>= 1)
#pragma unroll
      for (int r = 0; r < 8; ++r) tmax[r] = fmaxf(tmax[r], __shfl_xor(tmax[r], off, 32));

    v8f psum;
#pragma unroll
    for (int r = 0; r < 8; ++r) {
      float nm = fmaxf(rowM[r], tmax[r]);
      float corr = expf(rowM[r] - nm);
      rowM[r] = nm;
      rowL[r] *= corr;
      o[0][r] *= corr; o[1][r] *= corr; o[2][r] *= corr; o[3][r] *= corr;
      float pl = expf(s_lo[r] - nm);
      float ph = expf(s_hi[r] - nm);
      s_lo[r] = pl; s_hi[r] = ph;
      psum[r] = pl + ph;
    }
#pragma unroll
    for (int off = 8; off >= 1; off >>= 1)
#pragma unroll
      for (int r = 0; r < 8; ++r) psum[r] += __shfl_xor(psum[r], off, 32);
#pragma unroll
    for (int r = 0; r < 8; ++r) rowL[r] += psum[r];

    // stage P (16x32) to LDS in row-major f16, reload as an A-fragment.
#pragma unroll
    for (int r = 0; r < 8; ++r) {
      int row = half * 8 + r;
      pbuf[w][row][lane & 15]        = (_Float16)s_lo[r];
      pbuf[w][row][(lane & 15) + 16] = (_Float16)s_hi[r];
    }
    __builtin_amdgcn_wave_barrier();
    asm volatile("" ::: "memory");        // per-wave LDS ops are in-order (DScnt)
    {
      const _Float16* pr = &pbuf[w][lane & 15][(lane & 16) ? 8 : 0];
      v8h lo = *(const v8h*)pr;
      v8h hi = *(const v8h*)(pr + 16);
      v16h pa;
#pragma unroll
      for (int i = 0; i < 8; ++i) { pa[i] = lo[i]; pa[i + 8] = hi[i]; }
      o[0] = wmma16(pa, bfrag_T16(vb, n,  0, k0), o[0]);
      o[1] = wmma16(pa, bfrag_T16(vb, n, 16, k0), o[1]);
      o[2] = wmma16(pa, bfrag_T16(vb, n, 32, k0), o[2]);
      o[3] = wmma16(pa, bfrag_T16(vb, n, 48, k0), o[3]);
    }
    __builtin_amdgcn_wave_barrier();
    asm volatile("" ::: "memory");
  }

  // ---- memory-value contribution with final row maxima ----
#pragma unroll
  for (int m = 0; m < 16; ++m) {
    float mv = __shfl(rowM[m & 7], (m < 8) ? 0 : 16, 32);
    float p = (smem_s[m] > -1e29f) ? expf(smem_s[m] - mv) : 0.f;
    pmbuf[w][m][lane] = p;
  }
  __builtin_amdgcn_wave_barrier();
  asm volatile("" ::: "memory");
#pragma unroll
  for (int t = 0; t < 4; ++t)
#pragma unroll
    for (int r = 0; r < 8; ++r) {
      int row = half * 8 + r;
      int i = qRow0 + row;
      int col = t * 16 + (lane & 15);
      const float* mvp = mem_kv + (((size_t)bh * n + i) * KRET * 2 + 1) * dh + col;
      float acc = 0.f;
#pragma unroll 8
      for (int j = 0; j < KRET; ++j) acc = fmaf(pmbuf[w][row][j], mvp[(size_t)j * 2 * dh], acc);
      o[t][r] += acc;
    }

  // ---- normalize and emit attn_out (b, n, h*dh) f16 ----
#pragma unroll
  for (int t = 0; t < 4; ++t)
#pragma unroll
    for (int r = 0; r < 8; ++r) {
      int row = half * 8 + r;
      int i = qRow0 + row;
      int col = t * 16 + (lane & 15);
      float val = o[t][r] / rowL[r];
      attn_out[((size_t)bIdx * n + i) * 512 + hIdx * 64 + col] = (_Float16)val;
    }
}

// ---------------------------------------------------------------------------
// Kernel 4: out(f32) = attn_out(f16) @ Wout(f32) via WMMA.
// ---------------------------------------------------------------------------
__global__ __launch_bounds__(128)
void gemm_out_wmma(const _Float16* __restrict__ A, const float* __restrict__ B,
                   float* __restrict__ C, int K, int N) {
  int w = threadIdx.x >> 5;
  int lane = threadIdx.x & 31;
  int mt = blockIdx.x * 16;
  int nt = (blockIdx.y * 4 + w) * 16;
  v8f c = {};
  for (int k0 = 0; k0 < K; k0 += 32)
    c = wmma16(afrag_f16(A, K, mt, k0), bfrag_f32(B, N, k0, nt), c);
  int col = nt + (lane & 15);
  int rbase = (lane & 16) ? 8 : 0;
#pragma unroll
  for (int r = 0; r < 8; ++r)
    C[(size_t)(mt + rbase + r) * N + col] = c[r];
}

// ---------------------------------------------------------------------------
// Launch. Inputs: x, mem_kv, mem_mask, rel_pos_bias, Wq, Wkv, Wout, scale_param.
// ---------------------------------------------------------------------------
extern "C" void kernel_launch(void* const* d_in, const int* in_sizes, int n_in,
                              void* d_out, int out_size, void* d_ws, size_t ws_size,
                              hipStream_t stream) {
  const float*         x           = (const float*)d_in[0];
  const float*         mem_kv      = (const float*)d_in[1];
  const unsigned char* mem_mask    = (const unsigned char*)d_in[2];
  const float*         bias        = (const float*)d_in[3];
  const float*         Wq          = (const float*)d_in[4];
  const float*         Wkv         = (const float*)d_in[5];
  const float*         Wout        = (const float*)d_in[6];
  const float*         scale_param = (const float*)d_in[7];

  char* ws = (char*)d_ws;
  float*     qraw  = (float*)(ws + 0);            //  8 MB : x@Wq  (4096 x 512) f32
  float*     kvraw = (float*)(ws + 8388608);      //  2 MB : x@Wkv (4096 x 128) f32
  float*     qn    = (float*)(ws + 10485760);     //  8 MB : normalized+scaled q f32 (b,h,n,dh)
  _Float16*  qh    = (_Float16*)(ws + 18874368);  //  4 MB : same in f16
  _Float16*  kf    = (_Float16*)(ws + 23068672);  // .5 MB : normalized k f16 (b,n,dh)
  _Float16*  vT    = (_Float16*)(ws + 23592960);  // .5 MB : v transposed f16 (b,dh,n)
  _Float16*  aout  = (_Float16*)(ws + 24117248);  //  4 MB : attention out f16 (b*n, 512)
  float*     out   = (float*)d_out;

  // M = b*n = 4096 rows -> grid.x = 256 tiles of 16.
  gemm_f32_wmma<<<dim3(256, 8), 128, 0, stream>>>(x, Wq,  qraw,  512, 512);
  gemm_f32_wmma<<<dim3(256, 2), 128, 0, stream>>>(x, Wkv, kvraw, 512, 128);
  norm_q_kernel<<<4096, 256, 0, stream>>>(qraw, scale_param, qn, qh);   // 32768 row-heads
  norm_kv_kernel<<<512, 256, 0, stream>>>(kvraw, kf, vT);               // 4096 rows
  attn_kernel<<<dim3(32, 16), 128, 0, stream>>>(qh, qn, kf, vT, mem_kv, mem_mask, bias, aout);
  gemm_out_wmma<<<dim3(256, 8), 128, 0, stream>>>(aout, Wout, out, 512, 512);
}
